// PointNetSetAbstraction_46119358825117
// MI455X (gfx1250) — matrix operand order repaired and checked by reading, not compiled, not run on hardware
//
#include <hip/hip_runtime.h>

typedef __attribute__((ext_vector_type(16))) _Float16 v16h;
typedef __attribute__((ext_vector_type(8)))  float    v8f;

#define BATCH   16
#define NPTS    4096
#define FEATD   64
#define NPOINT  1024
#define NSAMPLE 32
#define RAD2    0.04f
#define NROWS   (BATCH * NPOINT * NSAMPLE)   // 524288 grouped rows
#define BN_EPS  1e-5f

// ---------------------------------------------------------------------------
// Swizzled "A-fragment-ready" f16 layout (CDNA5 WMMA 16-bit A matrix 16x32):
//   row tile t = row/16, rho = row%16, k-chunk kc = ch/32, kp = ch%32
//   b = (kp>>3)&1 ; lane = rho + 16*b ; h = (kp&7) + 8*(kp>>4)
//   offset = ((t*nkc + kc)*32 + lane)*16 + h
// A lane then loads its whole 16-half fragment contiguously (2x b128).
// ---------------------------------------------------------------------------
__device__ __forceinline__ size_t swz_off(int row, int ch, int nkc) {
    int t   = row >> 4;
    int rho = row & 15;
    int kc  = ch >> 5;
    int kp  = ch & 31;
    int lane = rho + (((kp >> 3) & 1) << 4);
    int h    = (kp & 7) + ((kp >> 4) << 3);
    return ((size_t)((t * nkc + kc) * 32 + lane)) * 16 + (size_t)h;
}

// ---------------------------------------------------------------------------
// Kernel 1: farthest point sampling. One workgroup (1024 thr = 32 waves) per
// batch. xyz fully LDS-resident; per-thread min-distances in registers;
// argmax via wave shuffles + 32-entry LDS tree. First-index tie-break to
// match jnp.argmax. Also emits new_xyz (first tuple output) directly.
// ---------------------------------------------------------------------------
__global__ __launch_bounds__(1024)
void fps_kernel(const float* __restrict__ pc, int* __restrict__ fps_idx,
                float* __restrict__ out_xyz) {
    const int b   = blockIdx.x;
    const int tid = threadIdx.x;
    const int lane = tid & 31, wv = tid >> 5;

    __shared__ float lx[NPTS], ly[NPTS], lz[NPTS];
    __shared__ float rv[32];
    __shared__ int   ri[32];
    __shared__ int   s_far;
    __shared__ int   s_fps[NPOINT];

    for (int i = tid; i < NPTS; i += 1024) {
        lx[i] = pc[(size_t)(b * 3 + 0) * NPTS + i];
        ly[i] = pc[(size_t)(b * 3 + 1) * NPTS + i];
        lz[i] = pc[(size_t)(b * 3 + 2) * NPTS + i];
    }
    __syncthreads();

    float d0 = 1e10f, d1 = 1e10f, d2 = 1e10f, d3 = 1e10f;
    int far = 0;

    for (int it = 0; it < NPOINT; ++it) {
        if (tid == 0) { fps_idx[b * NPOINT + it] = far; s_fps[it] = far; }
        const float cx = lx[far], cy = ly[far], cz = lz[far];

        float bv; int bi;
        {   // per-thread points: tid, tid+1024, tid+2048, tid+3072 (ascending)
            int n = tid;
            float dx = lx[n]-cx, dy = ly[n]-cy, dz = lz[n]-cz;
            d0 = fminf(d0, dx*dx + dy*dy + dz*dz);
            bv = d0; bi = n;
        }
        {   int n = tid + 1024;
            float dx = lx[n]-cx, dy = ly[n]-cy, dz = lz[n]-cz;
            d1 = fminf(d1, dx*dx + dy*dy + dz*dz);
            if (d1 > bv) { bv = d1; bi = n; }
        }
        {   int n = tid + 2048;
            float dx = lx[n]-cx, dy = ly[n]-cy, dz = lz[n]-cz;
            d2 = fminf(d2, dx*dx + dy*dy + dz*dz);
            if (d2 > bv) { bv = d2; bi = n; }
        }
        {   int n = tid + 3072;
            float dx = lx[n]-cx, dy = ly[n]-cy, dz = lz[n]-cz;
            d3 = fminf(d3, dx*dx + dy*dy + dz*dz);
            if (d3 > bv) { bv = d3; bi = n; }
        }
        // wave32 argmax reduction, lower index wins ties
        #pragma unroll
        for (int m = 1; m < 32; m <<= 1) {
            float ov = __shfl_xor(bv, m, 32);
            int   oi = __shfl_xor(bi, m, 32);
            if (ov > bv || (ov == bv && oi < bi)) { bv = ov; bi = oi; }
        }
        if (lane == 0) { rv[wv] = bv; ri[wv] = bi; }
        __syncthreads();
        if (tid == 0) {
            float best = rv[0]; int bidx = ri[0];
            for (int w = 1; w < 32; ++w)
                if (rv[w] > best || (rv[w] == best && ri[w] < bidx)) {
                    best = rv[w]; bidx = ri[w];
                }
            s_far = bidx;
        }
        __syncthreads();
        far = s_far;
    }

    // new_xyz -> d_out[(b*3+c)*NPOINT + s]
    for (int s = tid; s < NPOINT; s += 1024) {
        int idx = s_fps[s];
        out_xyz[(size_t)(b * 3 + 0) * NPOINT + s] = lx[idx];
        out_xyz[(size_t)(b * 3 + 1) * NPOINT + s] = ly[idx];
        out_xyz[(size_t)(b * 3 + 2) * NPOINT + s] = lz[idx];
    }
}

// ---------------------------------------------------------------------------
// Kernel 2: ball query. Reference's sort+where semantics == "first NSAMPLE
// in-radius indices with n < N-1 in ascending order, padded with the first
// hit (or N-1 if none)". One thread per centroid; all threads of a block
// scan the same batch's points in lockstep -> broadcast-friendly loads.
// ---------------------------------------------------------------------------
__global__ __launch_bounds__(256)
void ball_kernel(const float* __restrict__ pc, const int* __restrict__ fps_idx,
                 int* __restrict__ ball) {
    int g = blockIdx.x * blockDim.x + threadIdx.x;     // b*NPOINT + s
    if (g >= BATCH * NPOINT) return;
    int b = g >> 10;
    const float* px = pc + (size_t)(b * 3) * NPTS;
    const float* py = px + NPTS;
    const float* pz = py + NPTS;
    int ci = fps_idx[g];
    float cx = px[ci], cy = py[ci], cz = pz[ci];

    int* o = ball + (size_t)g * NSAMPLE;
    int cnt = 0;
    int first = NPTS - 1;
    for (int n = 0; n < NPTS - 1 && cnt < NSAMPLE; ++n) {
        float dx = px[n]-cx, dy = py[n]-cy, dz = pz[n]-cz;
        if (dx*dx + dy*dy + dz*dz <= RAD2) {
            if (cnt == 0) first = n;
            o[cnt++] = n;
        }
    }
    for (; cnt < NSAMPLE; ++cnt) o[cnt] = first;
}

// ---------------------------------------------------------------------------
// Kernel 3: build x0 = [grouped_xyz - centroid, grouped_feats] as f16 in the
// WMMA-A swizzled layout. Channels 0..2 coords, 3..66 feats, 67..95 zero pad.
// ---------------------------------------------------------------------------
__global__ __launch_bounds__(256)
void build_x0(const float* __restrict__ pc, const float* __restrict__ feats,
              const int* __restrict__ fps_idx, const int* __restrict__ ball,
              _Float16* __restrict__ x0) {
    int row = blockIdx.x * blockDim.x + threadIdx.x;
    if (row >= NROWS) return;
    int g  = row >> 5;          // b*NPOINT + s
    int b  = g >> 10;
    int n  = ball[row];
    int ci = fps_idx[g];
    const float* px = pc + (size_t)(b * 3) * NPTS;
    x0[swz_off(row, 0, 3)] = (_Float16)(px[n]          - px[ci]);
    x0[swz_off(row, 1, 3)] = (_Float16)(px[NPTS + n]   - px[NPTS + ci]);
    x0[swz_off(row, 2, 3)] = (_Float16)(px[2*NPTS + n] - px[2*NPTS + ci]);
    const float* f = feats + (size_t)b * FEATD * NPTS + n;
    #pragma unroll 4
    for (int c = 0; c < FEATD; ++c)
        x0[swz_off(row, 3 + c, 3)] = (_Float16)f[(size_t)c * NPTS];
    for (int c = 3 + FEATD; c < 96; ++c)
        x0[swz_off(row, c, 3)] = (_Float16)0.0f;
}

// ---------------------------------------------------------------------------
// Kernel 4: prepack weights into WMMA-B fragments (f16). B 32x16 layout:
// lane = col + 16*((kp>>3)&1), half h -> K' = (h&7) + 8*(lane>=16) + 16*(h>=8).
// ---------------------------------------------------------------------------
__global__ __launch_bounds__(256)
void prep_bfrag(const float* __restrict__ W, int O, int Cin, int nkc,
                _Float16* __restrict__ bf) {
    int i = blockIdx.x * blockDim.x + threadIdx.x;
    int total = (O / 16) * nkc * 512;
    if (i >= total) return;
    int h    = i & 15;
    int lane = (i >> 4) & 31;
    int blk  = i >> 9;                  // nt*nkc + kc
    int kc   = blk % nkc;
    int nt   = blk / nkc;
    int col  = lane & 15;
    int kp   = (h & 7) + ((lane >> 4) << 3) + ((h >> 3) << 4);
    int kappa = kc * 32 + kp;
    int o     = nt * 16 + col;
    float v = (kappa < Cin) ? W[(size_t)o * Cin + kappa] : 0.0f;
    bf[i] = (_Float16)v;
}

// ---------------------------------------------------------------------------
// Kernel 5: WMMA GEMM, fused with optional input affine+ReLU (folded BN of
// the previous layer), bias add, swizzled f16 output, and deterministic
// per-channel sum / sum-of-squares partials for this layer's BN.
// One wave computes one 16-row tile times all NT 16-col tiles, K unrolled.
// ---------------------------------------------------------------------------
template<int NKC, int NT, int OCH, int NKC_OUT, bool ACT>
__global__ __launch_bounds__(256)
void gemm_kernel(const _Float16* __restrict__ A, const _Float16* __restrict__ Bf,
                 const float* __restrict__ bias,
                 const float* __restrict__ ain, const float* __restrict__ cin,
                 _Float16* __restrict__ H, float* __restrict__ partials) {
    __shared__ float s_sum[8][OCH];
    __shared__ float s_sq[8][OCH];

    const int lane = threadIdx.x & 31;
    const int wv   = threadIdx.x >> 5;
    const int t    = blockIdx.x * 8 + wv;       // 16-row tile index

    v16h a[NKC];
    #pragma unroll
    for (int kc = 0; kc < NKC; ++kc)
        a[kc] = *(const v16h*)(A + ((size_t)(t * NKC + kc) * 32 + lane) * 16);

    if constexpr (ACT) {
        const int bb = (lane >> 4) & 1;
        #pragma unroll
        for (int kc = 0; kc < NKC; ++kc) {
            #pragma unroll
            for (int h = 0; h < 16; ++h) {
                int ch = kc * 32 + ((h & 7) | (bb << 3) | ((h >> 3) << 4));
                float v = (float)a[kc][h];
                v = fmaxf(fmaf(v, ain[ch], cin[ch]), 0.0f);
                a[kc][h] = (_Float16)v;
            }
        }
    }

    const int col16 = lane & 15;
    const int hi    = lane >> 4;
    const int rowbase = t * 16 + hi * 8;

    #pragma unroll
    for (int nt = 0; nt < NT; ++nt) {
        v8f c = {0.f, 0.f, 0.f, 0.f, 0.f, 0.f, 0.f, 0.f};
        #pragma unroll
        for (int kc = 0; kc < NKC; ++kc) {
            v16h bfr = *(const v16h*)(Bf + ((size_t)(nt * NKC + kc) * 32 + lane) * 16);
            c = __builtin_amdgcn_wmma_f32_16x16x32_f16(
                    false, a[kc], false, bfr, (short)0, c, false, false);
        }
        const int   col = nt * 16 + col16;
        const float bv  = bias[col];
        float ps = 0.f, pq = 0.f;
        #pragma unroll
        for (int r = 0; r < 8; ++r) {
            float v = c[r] + bv;
            ps += v; pq += v * v;
            H[swz_off(rowbase + r, col, NKC_OUT)] = (_Float16)v;
        }
        // deterministic pair-sum across lanes L / L+16 (same column)
        ps += __shfl_xor(ps, 16, 32);
        pq += __shfl_xor(pq, 16, 32);
        if (hi == 0) { s_sum[wv][col] = ps; s_sq[wv][col] = pq; }
    }
    __syncthreads();
    for (int ch = threadIdx.x; ch < OCH; ch += 256) {
        float s = 0.f, q = 0.f;
        #pragma unroll
        for (int w = 0; w < 8; ++w) { s += s_sum[w][ch]; q += s_sq[w][ch]; }
        partials[(size_t)blockIdx.x * (2 * OCH) + ch]       = s;
        partials[(size_t)blockIdx.x * (2 * OCH) + OCH + ch] = q;
    }
}

// ---------------------------------------------------------------------------
// Kernel 6: fixed-order reduction of BN partials -> folded affine (a, c):
//   y = relu(a*h + c) with a = gamma*rsqrt(var+eps), c = beta - mu*a
// ---------------------------------------------------------------------------
__global__ void finalize_stats(const float* __restrict__ partials, int nblocks,
                               int OCH, const float* __restrict__ gamma,
                               const float* __restrict__ beta,
                               float* __restrict__ aout, float* __restrict__ cout) {
    int ch = blockIdx.x * blockDim.x + threadIdx.x;
    if (ch >= OCH) return;
    float s = 0.f, q = 0.f;
    for (int bk = 0; bk < nblocks; ++bk) {
        s += partials[(size_t)bk * (2 * OCH) + ch];
        q += partials[(size_t)bk * (2 * OCH) + OCH + ch];
    }
    const float invC = 1.0f / (float)NROWS;
    float mu  = s * invC;
    float var = q * invC - mu * mu;
    float sc  = gamma[ch] * rsqrtf(var + BN_EPS);
    aout[ch] = sc;
    cout[ch] = beta[ch] - mu * sc;
}

// ---------------------------------------------------------------------------
// Kernel 7: fused BN + ReLU + max over the 32 neighbors; writes the
// transposed (B, 128, NPOINT) feature output with coalesced stores.
// ---------------------------------------------------------------------------
__global__ __launch_bounds__(256)
void maxpool_kernel(const _Float16* __restrict__ H2, const float* __restrict__ a2,
                    const float* __restrict__ c2, float* __restrict__ out) {
    int g = blockIdx.x * blockDim.x + threadIdx.x;     // (b*128+ch)*NPOINT + s
    if (g >= BATCH * 128 * NPOINT) return;
    int s   = g & (NPOINT - 1);
    int tmp = g >> 10;
    int ch  = tmp & 127;
    int b   = tmp >> 7;
    int rowbase = (b * NPOINT + s) << 5;
    float aa = a2[ch], cc = c2[ch];
    float m = 0.0f;                                   // relu output >= 0
    #pragma unroll 8
    for (int k = 0; k < NSAMPLE; ++k) {
        float v = (float)H2[swz_off(rowbase + k, ch, 4)];
        m = fmaxf(m, fmaxf(fmaf(v, aa, cc), 0.0f));
    }
    out[g] = m;
}

// ---------------------------------------------------------------------------
extern "C" void kernel_launch(void* const* d_in, const int* in_sizes, int n_in,
                              void* d_out, int out_size, void* d_ws, size_t ws_size,
                              hipStream_t stream) {
    (void)in_sizes; (void)n_in; (void)out_size; (void)ws_size;

    const float* pc    = (const float*)d_in[0];
    const float* feats = (const float*)d_in[1];
    const float* W0 = (const float*)d_in[2];
    const float* b0 = (const float*)d_in[3];
    const float* g0 = (const float*)d_in[4];
    const float* e0 = (const float*)d_in[5];
    const float* W1 = (const float*)d_in[6];
    const float* b1 = (const float*)d_in[7];
    const float* g1 = (const float*)d_in[8];
    const float* e1 = (const float*)d_in[9];
    const float* W2 = (const float*)d_in[10];
    const float* b2 = (const float*)d_in[11];
    const float* g2 = (const float*)d_in[12];
    const float* e2 = (const float*)d_in[13];

    char* ws = (char*)d_ws;
    size_t off = 0;
    auto alloc = [&](size_t bytes) {
        size_t o = off;
        off = (off + bytes + 255) & ~(size_t)255;
        return o;
    };

    int*       fps  = (int*)      (ws + alloc((size_t)BATCH * NPOINT * 4));
    int*       ball = (int*)      (ws + alloc((size_t)NROWS * 4));
    _Float16*  x0   = (_Float16*) (ws + alloc((size_t)NROWS * 96  * 2));
    _Float16*  h0   = (_Float16*) (ws + alloc((size_t)NROWS * 64  * 2));
    _Float16*  h1   = (_Float16*) (ws + alloc((size_t)NROWS * 64  * 2));
    _Float16*  h2   = (_Float16*) (ws + alloc((size_t)NROWS * 128 * 2));
    _Float16*  bf0  = (_Float16*) (ws + alloc((size_t)4 * 3 * 512 * 2));
    _Float16*  bf1  = (_Float16*) (ws + alloc((size_t)4 * 2 * 512 * 2));
    _Float16*  bf2  = (_Float16*) (ws + alloc((size_t)8 * 2 * 512 * 2));
    float*     part = (float*)    (ws + alloc((size_t)4096 * 256 * 4));
    float*     aff  = (float*)    (ws + alloc((size_t)6 * 128 * 4));
    float* a0 = aff;        float* c0 = aff + 128;
    float* a1 = aff + 256;  float* c1 = aff + 384;
    float* a2 = aff + 512;  float* c2 = aff + 640;

    float* newxyz_out = (float*)d_out;                              // (16,3,1024)
    float* feat_out   = (float*)d_out + (size_t)BATCH * 3 * NPOINT; // (16,128,1024)

    // weight prepack (tiny)
    prep_bfrag<<<(6144 + 255) / 256, 256, 0, stream>>>(W0, 64, 67, 3, bf0);
    prep_bfrag<<<(4096 + 255) / 256, 256, 0, stream>>>(W1, 64, 64, 2, bf1);
    prep_bfrag<<<(8192 + 255) / 256, 256, 0, stream>>>(W2, 128, 64, 2, bf2);

    // sampling / grouping
    fps_kernel <<<BATCH, 1024, 0, stream>>>(pc, fps, newxyz_out);
    ball_kernel<<<(BATCH * NPOINT) / 256, 256, 0, stream>>>(pc, fps, ball);
    build_x0   <<<NROWS / 256, 256, 0, stream>>>(pc, feats, fps, ball, x0);

    // MLP: 3x WMMA GEMM with fused BN folding
    gemm_kernel<3, 4, 64, 2, false><<<4096, 256, 0, stream>>>(x0, bf0, b0, a0, c0, h0, part);
    finalize_stats<<<1, 64, 0, stream>>>(part, 4096, 64, g0, e0, a0, c0);

    gemm_kernel<2, 4, 64, 2, true><<<4096, 256, 0, stream>>>(h0, bf1, b1, a0, c0, h1, part);
    finalize_stats<<<1, 64, 0, stream>>>(part, 4096, 64, g1, e1, a1, c1);

    gemm_kernel<2, 8, 128, 4, true><<<4096, 256, 0, stream>>>(h1, bf2, b2, a1, c1, h2, part);
    finalize_stats<<<1, 128, 0, stream>>>(part, 4096, 128, g2, e2, a2, c2);

    // fused BN + ReLU + max-pool + transpose
    maxpool_kernel<<<(BATCH * 128 * NPOINT) / 256, 256, 0, stream>>>(h2, a2, c2, feat_out);
}
